// PIGNN_91001767068115
// MI455X (gfx1250) — compile-verified, matrix operand order checked by reading, not run on hardware
//
#include <hip/hip_runtime.h>
#include <hip/hip_bf16.h>

#define HID 64

typedef __attribute__((ext_vector_type(16))) _Float16 v16h;
typedef __attribute__((ext_vector_type(8)))  float    v8f;

union Frag16 {
  v16h h;
  uint4 u[2];
};

__device__ __forceinline__ v8f wmma_f16(v16h a, v16h b, v8f c) {
  // D = A(16x32 f16) x B(32x16 f16) + C(16x16 f32)
  return __builtin_amdgcn_wmma_f32_16x16x32_f16(
      /*neg_a=*/false, a, /*neg_b=*/false, b,
      /*c_mod=*/(short)0, c, /*reuse_a=*/false, /*reuse_b=*/false);
}

// Gather one 8-half (16B) chunk of the concatenated edge feature row
// [ h[dst] (64) | h[src] (64) | edge_attr (4) | pad -> 160 ].
__device__ __forceinline__ uint4 gather_chunk(const uint4* hD, const uint4* hS,
                                              const float* __restrict__ attr,
                                              long e, int kstart) {
  if (kstart < 64)  return hD[kstart >> 3];
  if (kstart < 128) return hS[(kstart - 64) >> 3];
  union { uint4 u; _Float16 h[8]; } p;
  p.u = make_uint4(0u, 0u, 0u, 0u);
  if (kstart == 128) {
    const float4 a4 = *(const float4*)(attr + e * 4);
    p.h[0] = (_Float16)a4.x; p.h[1] = (_Float16)a4.y;
    p.h[2] = (_Float16)a4.z; p.h[3] = (_Float16)a4.w;
  }
  return p.u;
}

// Convert one 8-float chunk of [ h (64 fp32) | agg (64 fp32) ] to 8 halves.
__device__ __forceinline__ uint4 cvt_chunk(const float* __restrict__ hRow,
                                           const float* __restrict__ aRow,
                                           int kstart) {
  const float* s = (kstart < 64) ? (hRow + kstart) : (aRow + (kstart - 64));
  float4 f0 = *(const float4*)s;
  float4 f1 = *(const float4*)(s + 4);
  union { uint4 u; _Float16 h[8]; } p;
  p.h[0] = (_Float16)f0.x; p.h[1] = (_Float16)f0.y;
  p.h[2] = (_Float16)f0.z; p.h[3] = (_Float16)f0.w;
  p.h[4] = (_Float16)f1.x; p.h[5] = (_Float16)f1.y;
  p.h[6] = (_Float16)f1.z; p.h[7] = (_Float16)f1.w;
  return p.u;
}

// ---------------------------------------------------------------- embed ----
__global__ __launch_bounds__(256) void embed_kernel(
    const float* __restrict__ x, const float* __restrict__ W,
    const float* __restrict__ b, float* __restrict__ h,
    _Float16* __restrict__ hh, long total) {
  long i = (long)blockIdx.x * blockDim.x + threadIdx.x;
  if (i >= total) return;
  int  j = (int)(i & 63);
  long n = i >> 6;
  const float* xr = x + n * 10;
  float acc = b[j];
#pragma unroll
  for (int k = 0; k < 10; ++k) acc += xr[k] * W[k * HID + j];
  h[i]  = acc;
  hh[i] = (_Float16)acc;
}

// ----------------------------------------------------------------- zero ----
__global__ __launch_bounds__(256) void zero_kernel(float4* __restrict__ p, long n4) {
  long i = (long)blockIdx.x * blockDim.x + threadIdx.x;
  long stride = (long)gridDim.x * blockDim.x;
  const float4 z = make_float4(0.f, 0.f, 0.f, 0.f);
  for (; i < n4; i += stride) p[i] = z;
}

// --------------------------------------------------------- edge MLP + scatter
// Persistent blocks: fill LDS weights once, then grid-stride over 128-edge
// tiles (8 waves x 16 edges). Amortizes the ~28KB LDS weight stage across
// many tiles instead of paying it per 128 edges.
__global__ __launch_bounds__(256) void edge_mlp_kernel(
    const _Float16* __restrict__ hh,
    const int* __restrict__ srcIdx, const int* __restrict__ dstIdx,
    const float* __restrict__ attr,
    const float* __restrict__ W1, const float* __restrict__ B1,
    const float* __restrict__ W2, const float* __restrict__ B2,
    float* __restrict__ agg, int E, int numTiles) {
  __shared__ _Float16 sW1[64 * 160];      // em1_W transposed [n][k], K padded 132->160
  __shared__ _Float16 sW2[64 * 64];       // em2_W transposed [n][k]
  __shared__ _Float16 sM [8 * 16 * 64];   // per-wave layer-1 output (transpose bounce)

  const int t = threadIdx.x;
  for (int i = t; i < 64 * 160; i += 256) {
    int n = i / 160, k = i % 160;
    sW1[i] = (k < 132) ? (_Float16)W1[k * HID + n] : (_Float16)0.0f;
  }
  for (int i = t; i < 64 * 64; i += 256) {
    int n = i >> 6, k = i & 63;
    sW2[i] = (_Float16)W2[k * HID + n];
  }
  __syncthreads();

  const int wave  = t >> 5;
  const int lane  = t & 31;
  const int ln15  = lane & 15;            // A row / B,C column
  const int koff  = (lane >> 4) << 3;     // 0 or 8
  const int base8 = (lane >> 4) << 3;     // C/D row base (M offset 0 or 8)
  _Float16* mysc = sM + wave * (16 * 64);

  // bias registers are tile-invariant: load once
  float biasL1[4], biasL2[4];
#pragma unroll
  for (int nt = 0; nt < 4; ++nt) {
    biasL1[nt] = B1[nt * 16 + ln15];
    biasL2[nt] = B2[nt * 16 + ln15];
  }

  for (long tile = blockIdx.x; tile < numTiles; tile += gridDim.x) {
    const long tileBase = tile * 128 + wave * 16;

    // prefetch next tile's index cachelines while this tile computes
    long nextE = (tile + gridDim.x) * 128 + wave * 16;
    if (nextE < E) {
      __builtin_prefetch(dstIdx + nextE, 0, 0);
      __builtin_prefetch(srcIdx + nextE, 0, 0);
    }

    long e = tileBase + ln15;
    if (e >= E) e = E - 1;                // clamp gathers; scatter is guarded
    const int di = dstIdx[e];
    const int si = srcIdx[e];
    const uint4* hD = (const uint4*)(hh + (size_t)di * HID);
    const uint4* hS = (const uint4*)(hh + (size_t)si * HID);

    // ---- layer 1: [16,160] x [160,64] -> [16,64], fp32 acc
    v8f acc[4] = {};
#pragma unroll
    for (int kb = 0; kb < 5; ++kb) {
      Frag16 a;
      a.u[0] = gather_chunk(hD, hS, attr, e, kb * 32 + koff);
      a.u[1] = gather_chunk(hD, hS, attr, e, kb * 32 + 16 + koff);
#pragma unroll
      for (int nt = 0; nt < 4; ++nt) {
        Frag16 b;
        const uint4* wrow = (const uint4*)(sW1 + (nt * 16 + ln15) * 160);
        b.u[0] = wrow[(kb * 32 + koff) >> 3];
        b.u[1] = wrow[(kb * 32 + 16 + koff) >> 3];
        acc[nt] = wmma_f16(a.h, b.h, acc[nt]);
      }
    }

    // ---- bias + relu, bounce through per-wave LDS (C/D layout -> A layout)
#pragma unroll
    for (int nt = 0; nt < 4; ++nt) {
#pragma unroll
      for (int v = 0; v < 8; ++v) {
        float xv = acc[nt][v] + biasL1[nt];
        xv = xv > 0.0f ? xv : 0.0f;
        mysc[(base8 + v) * HID + nt * 16 + ln15] = (_Float16)xv;
      }
    }
    asm volatile("s_wait_dscnt 0" ::: "memory");  // wave-local LDS RAW

    // ---- layer 2: [16,64] x [64,64] -> [16,64]
    v8f acc2[4] = {};
    const uint4* arow = (const uint4*)(mysc + ln15 * HID);
#pragma unroll
    for (int kb = 0; kb < 2; ++kb) {
      Frag16 a;
      a.u[0] = arow[(kb * 32 + koff) >> 3];
      a.u[1] = arow[(kb * 32 + 16 + koff) >> 3];
#pragma unroll
      for (int nt = 0; nt < 4; ++nt) {
        Frag16 b;
        const uint4* wrow = (const uint4*)(sW2 + (nt * 16 + ln15) * HID);
        b.u[0] = wrow[(kb * 32 + koff) >> 3];
        b.u[1] = wrow[(kb * 32 + 16 + koff) >> 3];
        acc2[nt] = wmma_f16(a.h, b.h, acc2[nt]);
      }
    }

    // ---- bias + relu + scatter-add into agg[dst]
    int dstOf[8];
#pragma unroll
    for (int v = 0; v < 8; ++v) {
      long ee = tileBase + base8 + v;
      dstOf[v] = (ee < E) ? dstIdx[ee] : -1;
    }
#pragma unroll
    for (int nt = 0; nt < 4; ++nt) {
#pragma unroll
      for (int v = 0; v < 8; ++v) {
        if (dstOf[v] >= 0) {
          float xv = acc2[nt][v] + biasL2[nt];
          xv = xv > 0.0f ? xv : 0.0f;
          atomicAdd(&agg[(size_t)dstOf[v] * HID + nt * 16 + ln15], xv);
        }
      }
    }
  }
}

// ------------------------------------------------------------ node update --
// Persistent blocks; wave owns a 16-node M-tile (128 nodes/block-iteration).
__global__ __launch_bounds__(256) void node_mlp_kernel(
    float* __restrict__ h, const float* __restrict__ agg,
    const float* __restrict__ W1, const float* __restrict__ B1,
    const float* __restrict__ W2, const float* __restrict__ B2,
    _Float16* __restrict__ hh, long Nn, int numTiles) {
  __shared__ _Float16 sW1[64 * 128];      // nu1_W transposed [n][k]
  __shared__ _Float16 sW2[64 * 64];       // nu2_W transposed [n][k]
  __shared__ _Float16 sM [8 * 16 * 64];

  const int t = threadIdx.x;
  for (int i = t; i < 64 * 128; i += 256) {
    int n = i >> 7, k = i & 127;
    sW1[i] = (_Float16)W1[k * HID + n];
  }
  for (int i = t; i < 64 * 64; i += 256) {
    int n = i >> 6, k = i & 63;
    sW2[i] = (_Float16)W2[k * HID + n];
  }
  __syncthreads();

  const int wave  = t >> 5;
  const int lane  = t & 31;
  const int ln15  = lane & 15;
  const int koff  = (lane >> 4) << 3;
  const int base8 = (lane >> 4) << 3;
  _Float16* mysc = sM + wave * (16 * 64);

  float biasL1[4], biasL2[4];
#pragma unroll
  for (int nt = 0; nt < 4; ++nt) {
    biasL1[nt] = B1[nt * 16 + ln15];
    biasL2[nt] = B2[nt * 16 + ln15];
  }

  for (long tile = blockIdx.x; tile < numTiles; tile += gridDim.x) {
    const long tileBase = tile * 128 + wave * 16;

    long nd = tileBase + ln15;
    if (nd >= Nn) nd = Nn - 1;
    const float* hRow = h   + (size_t)nd * HID;
    const float* aRow = agg + (size_t)nd * HID;

    // ---- layer 1: [16,128] x [128,64]
    v8f acc[4] = {};
#pragma unroll
    for (int kb = 0; kb < 4; ++kb) {
      Frag16 a;
      a.u[0] = cvt_chunk(hRow, aRow, kb * 32 + koff);
      a.u[1] = cvt_chunk(hRow, aRow, kb * 32 + 16 + koff);
#pragma unroll
      for (int nt = 0; nt < 4; ++nt) {
        Frag16 b;
        const uint4* wrow = (const uint4*)(sW1 + (nt * 16 + ln15) * 128);
        b.u[0] = wrow[(kb * 32 + koff) >> 3];
        b.u[1] = wrow[(kb * 32 + 16 + koff) >> 3];
        acc[nt] = wmma_f16(a.h, b.h, acc[nt]);
      }
    }

#pragma unroll
    for (int nt = 0; nt < 4; ++nt) {
#pragma unroll
      for (int v = 0; v < 8; ++v) {
        float xv = acc[nt][v] + biasL1[nt];
        xv = xv > 0.0f ? xv : 0.0f;
        mysc[(base8 + v) * HID + nt * 16 + ln15] = (_Float16)xv;
      }
    }
    asm volatile("s_wait_dscnt 0" ::: "memory");

    // ---- layer 2: [16,64] x [64,64]
    v8f acc2[4] = {};
    const uint4* arow = (const uint4*)(mysc + ln15 * HID);
#pragma unroll
    for (int kb = 0; kb < 2; ++kb) {
      Frag16 a;
      a.u[0] = arow[(kb * 32 + koff) >> 3];
      a.u[1] = arow[(kb * 32 + 16 + koff) >> 3];
#pragma unroll
      for (int nt = 0; nt < 4; ++nt) {
        Frag16 b;
        const uint4* wrow = (const uint4*)(sW2 + (nt * 16 + ln15) * HID);
        b.u[0] = wrow[(kb * 32 + koff) >> 3];
        b.u[1] = wrow[(kb * 32 + 16 + koff) >> 3];
        acc2[nt] = wmma_f16(a.h, b.h, acc2[nt]);
      }
    }

    // ---- bias + relu; write h (fp32, in place) and f16 mirror
#pragma unroll
    for (int nt = 0; nt < 4; ++nt) {
#pragma unroll
      for (int v = 0; v < 8; ++v) {
        long node = tileBase + base8 + v;
        if (node < Nn) {
          float xv = acc2[nt][v] + biasL2[nt];
          xv = xv > 0.0f ? xv : 0.0f;
          h [(size_t)node * HID + nt * 16 + ln15] = xv;
          hh[(size_t)node * HID + nt * 16 + ln15] = (_Float16)xv;
        }
      }
    }
  }
}

// ---------------------------------------------------------------- reduce ---
__global__ __launch_bounds__(256) void reduce_kernel(
    const float* __restrict__ h, float* __restrict__ accum, long Nn) {
  __shared__ float sred[256];
  const int j = threadIdx.x & 63;
  const int g = threadIdx.x >> 6;  // 4 node-groups per block
  float s = 0.0f;
  for (long n = (long)blockIdx.x * 4 + g; n < Nn; n += (long)gridDim.x * 4)
    s += h[n * HID + j];
  sred[threadIdx.x] = s;
  __syncthreads();
  if (g == 0) {
    float tot = sred[j] + sred[64 + j] + sred[128 + j] + sred[192 + j];
    atomicAdd(&accum[j], tot);
  }
}

// ------------------------------------------------------------------ head ---
__global__ __launch_bounds__(64) void head_kernel(
    const float* __restrict__ accum,
    const float* __restrict__ W1, const float* __restrict__ B1,
    const float* __restrict__ W2, const float* __restrict__ B2,
    float* __restrict__ out, float invN) {
  __shared__ float hg[64];
  __shared__ float t1[64];
  const int j = threadIdx.x;
  hg[j] = accum[j] * invN;
  __syncthreads();
  float a = B1[j];
#pragma unroll
  for (int k = 0; k < 64; ++k) a += hg[k] * W1[k * HID + j];
  t1[j] = a > 0.0f ? a : 0.0f;
  __syncthreads();
  if (j < 3) {
    float o = B2[j];
#pragma unroll
    for (int k = 0; k < 64; ++k) o += t1[k] * W2[k * 3 + j];
    out[j] = o;
  }
}

// ---------------------------------------------------------------- launch ---
extern "C" void kernel_launch(void* const* d_in, const int* in_sizes, int n_in,
                              void* d_out, int out_size, void* d_ws, size_t ws_size,
                              hipStream_t stream) {
  const float* x     = (const float*)d_in[0];
  const int*   eidx  = (const int*)  d_in[1];
  const float* attr  = (const float*)d_in[2];
  const float* embW  = (const float*)d_in[3];
  const float* embB  = (const float*)d_in[4];
  const float* em1W  = (const float*)d_in[5];
  const float* em1B  = (const float*)d_in[6];
  const float* em2W  = (const float*)d_in[7];
  const float* em2B  = (const float*)d_in[8];
  const float* nu1W  = (const float*)d_in[9];
  const float* nu1B  = (const float*)d_in[10];
  const float* nu2W  = (const float*)d_in[11];
  const float* nu2B  = (const float*)d_in[12];
  const float* hd1W  = (const float*)d_in[13];
  const float* hd1B  = (const float*)d_in[14];
  const float* hd2W  = (const float*)d_in[15];
  const float* hd2B  = (const float*)d_in[16];

  const long Nn = in_sizes[0] / 10;
  const long E  = in_sizes[1] / 2;
  const int* src = eidx;        // edge_index[0]
  const int* dst = eidx + E;    // edge_index[1]

  // workspace layout (256B-aligned slices)
  auto align256 = [](size_t v) { return (v + 255) & ~(size_t)255; };
  char* ws = (char*)d_ws;
  size_t szH  = align256((size_t)Nn * HID * sizeof(float));
  size_t szHH = align256((size_t)Nn * HID * sizeof(_Float16));
  float*    h     = (float*)   (ws);
  _Float16* hh    = (_Float16*)(ws + szH);
  float*    agg   = (float*)   (ws + szH + szHH);
  float*    accum = (float*)   (ws + szH + szHH + szH);

  const long total = Nn * HID;

  embed_kernel<<<(int)((total + 255) / 256), 256, 0, stream>>>(
      x, embW, embB, h, hh, total);

  const int edgeTiles = (int)((E + 127) / 128);
  const int nodeTiles = (int)((Nn + 127) / 128);
  const int edgeGrid  = edgeTiles < 2048 ? edgeTiles : 2048;
  const int nodeGrid  = nodeTiles < 1024 ? nodeTiles : 1024;

  for (int step = 0; step < 2; ++step) {
    zero_kernel<<<1024, 256, 0, stream>>>((float4*)agg, total / 4);
    edge_mlp_kernel<<<edgeGrid, 256, 0, stream>>>(
        hh, src, dst, attr, em1W, em1B, em2W, em2B, agg, (int)E, edgeTiles);
    node_mlp_kernel<<<nodeGrid, 256, 0, stream>>>(
        h, agg, nu1W, nu1B, nu2W, nu2B, hh, Nn, nodeTiles);
  }

  zero_kernel<<<1, 16, 0, stream>>>((float4*)accum, 16);
  reduce_kernel<<<256, 256, 0, stream>>>(h, accum, Nn);
  head_kernel<<<1, 64, 0, stream>>>(
      accum, hd1W, hd1B, hd2W, hd2B, (float*)d_out, 1.0f / (float)Nn);
}
